// LinearSelfAttentionBlock_80736795230453
// MI455X (gfx1250) — matrix operand order repaired and checked by reading, not compiled, not run on hardware
//
#include <hip/hip_runtime.h>
#include <hip/hip_bf16.h>

typedef __attribute__((ext_vector_type(16))) _Float16 v16h;
typedef __attribute__((ext_vector_type(8)))  _Float16 v8h;
typedef __attribute__((ext_vector_type(8)))  float    v8f;

static constexpr int Bb  = 4;
static constexpr int Cc  = 512;
static constexpr int Ll  = 4096;
static constexpr int Hh  = 8;
static constexpr int DH  = 64;
static constexpr int MLPd= 2048;
static constexpr int Nrow= Bb*Ll;          // 16384

// ---------------------------------------------------------------------------
// Tiled WMMA GEMM: out = epilogue(A[MxK] @ B[KxN])
// 256 threads = 8 waves; tile 128(M) x BN(N); K-step 32.
// A staged via GLOBAL_LOAD_ASYNC_TO_LDS_B128 (no VGPR round-trip, ASYNCcnt).
// B staged TRANSPOSED ([col][k]) through registers so fragments are ds_load_b128.
// Double-buffered LDS, one barrier per K-step.
// EPI: 0=+bias, 1=tanh(+bias)+1, 2=gelu(+bias), 3=divide-by-rowvec, 4=+bias+addend
// ---------------------------------------------------------------------------
template<int EPI, int BN, int WM>
__global__ __launch_bounds__(256, 1)
void gemm_wmma(const _Float16* __restrict__ A, const _Float16* __restrict__ Bw,
               float* __restrict__ outF, _Float16* __restrict__ outH,
               const float* __restrict__ bias, const float* __restrict__ addend,
               const float* __restrict__ rowdiv,
               int lda, int ldb, int ldo, int M, int K, int HH,
               long long sAb, long long sAh, long long sBb, long long sBh,
               long long sOb, long long sOh, long long sRz)
{
    constexpr int WN  = 8 / WM;          // waves along N
    constexpr int XA  = 128 / (16 * WM); // A-frags per wave
    constexpr int YB  = BN  / (16 * WN); // B-frags per wave
    constexpr int CBM = BN / 8 - 1;      // B col-chunk mask
    constexpr int CBS = (BN / 8 == 16) ? 4 : 3;

    const int z  = blockIdx.z;
    const int zb = z / HH, zh = z % HH;
    const _Float16* Ap = A  + zb*sAb + zh*sAh;
    const _Float16* Bp = Bw + zb*sBb + zh*sBh;
    const long long obase = (long long)zb*sOb + (long long)zh*sOh;
    const float* rd = rowdiv ? (rowdiv + (long long)z*sRz) : nullptr;

    const int tid  = threadIdx.x;
    const int wave = tid >> 5;
    const int lane = tid & 31;
    const int wm   = wave / WN;          // 0..WM-1
    const int wn   = wave % WN;          // 0..WN-1
    const int rowBase = blockIdx.x * 128;
    const int colBase = blockIdx.y * BN;

    __shared__ __align__(16) _Float16 sA [2][128 * 40];  // [row][k] (32 + 8 pad)
    __shared__ __align__(16) _Float16 sBt[2][BN  * 40];  // [col][k] (32 + 8 pad)

    v8f acc[XA][YB] = {};

    const int mrow = lane & 15;
    const int half = lane >> 4;

    uint4 rb[2];

    // A: async copy global -> LDS, 128 rows x 4 chunks of 16B, 2 per thread
    auto asyncA = [&](int kk, int buf) {
        #pragma unroll
        for (int i = 0; i < 2; ++i) {
            int id = tid + i * 256;
            int r = id >> 2, c = id & 3;
            const void* gp = Ap + (long long)(rowBase + r)*lda + kk + c*8;
            unsigned int ldsoff =
                (unsigned int)(unsigned long long)(void*)&sA[buf][r*40 + c*8];
            asm volatile("global_load_async_to_lds_b128 %0, %1, off"
                         :: "v"(ldsoff), "v"(gp) : "memory");
        }
    };
    auto loadB = [&](int kk) {
        // B: 16 row-pairs x BN/8 col-chunks = BN*2 tasks (<= 256)
        if (tid < BN * 2) {
            int cb = tid & CBM, rp = tid >> CBS;
            rb[0] = *reinterpret_cast<const uint4*>(
                Bp + (long long)(kk + 2*rp    )*ldb + colBase + cb*8);
            rb[1] = *reinterpret_cast<const uint4*>(
                Bp + (long long)(kk + 2*rp + 1)*ldb + colBase + cb*8);
        }
    };
    auto storeB = [&](int buf) {
        if (tid < BN * 2) {
            int cb = tid & CBM, rp = tid >> CBS;
            const unsigned short* lo = reinterpret_cast<const unsigned short*>(&rb[0]);
            const unsigned short* hi = reinterpret_cast<const unsigned short*>(&rb[1]);
            #pragma unroll
            for (int j = 0; j < 8; ++j) {
                unsigned int pk = (unsigned int)lo[j] | ((unsigned int)hi[j] << 16);
                *reinterpret_cast<unsigned int*>(&sBt[buf][(cb*8 + j)*40 + 2*rp]) = pk;
            }
        }
    };

    const int nk = K >> 5;

    asyncA(0, 0);
    loadB(0);
    storeB(0);
    asm volatile("s_wait_asynccnt 0x0" ::: "memory");
    __syncthreads();
    if (nk > 1) loadB(32);

    for (int t = 0; t < nk; ++t) {
        const int buf = t & 1;

        if (t + 1 < nk) asyncA((t + 1) << 5, (t + 1) & 1);  // dest buffer is free

        // A fragments: lanes 0-15 -> K {0..7,16..23}, lanes 16-31 -> K {8..15,24..31}
        const int ka = half * 8;
        v16h af[XA];
        #pragma unroll
        for (int am = 0; am < XA; ++am) {
            const int arow = wm*(XA*16) + am*16 + mrow;
            v8h a0 = *reinterpret_cast<const v8h*>(&sA[buf][arow*40 + ka]);
            v8h a1 = *reinterpret_cast<const v8h*>(&sA[buf][arow*40 + 16 + ka]);
            af[am] = __builtin_shufflevector(a0, a1, 0,1,2,3,4,5,6,7,8,9,10,11,12,13,14,15);
        }

        // B fragments: column = wn*(YB*16) + nt*16 + (lane&15), lanes 0-15 -> K {0..7,16..23}, lanes 16-31 -> K {8..15,24..31}
        const int kb = half * 8;
        #pragma unroll
        for (int nt = 0; nt < YB; ++nt) {
            const int bcol = wn*(YB*16) + nt*16 + mrow;
            v8h b0 = *reinterpret_cast<const v8h*>(&sBt[buf][bcol*40 + kb]);
            v8h b1 = *reinterpret_cast<const v8h*>(&sBt[buf][bcol*40 + kb + 16]);
            v16h bf = __builtin_shufflevector(b0, b1, 0,1,2,3,4,5,6,7,8,9,10,11,12,13,14,15);
            #pragma unroll
            for (int am = 0; am < XA; ++am)
                acc[am][nt] = __builtin_amdgcn_wmma_f32_16x16x32_f16(
                    false, af[am], false, bf, (short)0, acc[am][nt], false, false);
        }

        if (t + 1 < nk) {
            storeB((t + 1) & 1);                            // B(t+1) into other buffer
            asm volatile("s_wait_asynccnt 0x0" ::: "memory"); // A(t+1) landed
        }
        __syncthreads();
        if (t + 2 < nk) loadB((t + 2) << 5);                // in flight under next compute
    }

    // epilogue: row = wm*(XA*16) + am*16 + (lane>>4)*8 + r, col = wn*(YB*16) + nt*16 + mrow
    #pragma unroll
    for (int am = 0; am < XA; ++am) {
        #pragma unroll
        for (int nt = 0; nt < YB; ++nt) {
            const int col = colBase + wn*(YB*16) + nt*16 + mrow;
            #pragma unroll
            for (int r = 0; r < 8; ++r) {
                const int row = rowBase + wm*(XA*16) + am*16 + half*8 + r;
                float v = acc[am][nt][r];
                if (EPI == 0 || EPI == 1 || EPI == 2 || EPI == 4) v += bias[col];
                if (EPI == 1) v = tanhf(v) + 1.0f;
                if (EPI == 2) v = 0.5f * v * (1.0f + erff(v * 0.70710678118654752f));
                if (EPI == 3) v = v / rd[row];
                const long long o = obase + (long long)row*ldo + col;
                if (EPI == 4) v += addend[o];
                if (outF) outF[o] = v;
                if (outH) outH[o] = (_Float16)v;
            }
        }
    }
}

// ---------------------------------------------------------------------------
// LayerNorm over rows of width 512; writes f32 and optional f16 copy.
// ---------------------------------------------------------------------------
__global__ __launch_bounds__(256)
void ln_rows(const float* __restrict__ in, const float* __restrict__ g,
             const float* __restrict__ be, float* __restrict__ outF,
             _Float16* __restrict__ outH)
{
    const int row = blockIdx.x, t = threadIdx.x;
    const float* r = in + (size_t)row * Cc;
    float x0 = r[t], x1 = r[t + 256];
    __shared__ float s1[256], s2[256];
    s1[t] = x0 + x1;
    s2[t] = x0*x0 + x1*x1;
    __syncthreads();
    for (int o = 128; o > 0; o >>= 1) {
        if (t < o) { s1[t] += s1[t+o]; s2[t] += s2[t+o]; }
        __syncthreads();
    }
    const float mu   = s1[0] * (1.0f / Cc);
    const float var  = s2[0] * (1.0f / Cc) - mu * mu;
    const float rstd = rsqrtf(var + 1e-5f);
    float y0 = (x0 - mu) * rstd * g[t]       + be[t];
    float y1 = (x1 - mu) * rstd * g[t + 256] + be[t + 256];
    outF[(size_t)row*Cc + t]       = y0;
    outF[(size_t)row*Cc + t + 256] = y1;
    if (outH) {
        outH[(size_t)row*Cc + t]       = (_Float16)y0;
        outH[(size_t)row*Cc + t + 256] = (_Float16)y1;
    }
}

// x [B,C,L] -> xh [B*L, C] (f16), LDS-tiled transpose
__global__ __launch_bounds__(256)
void xpose_in(const float* __restrict__ x, _Float16* __restrict__ xh)
{
    const int b = blockIdx.z, l0 = blockIdx.x*32, c0 = blockIdx.y*32;
    __shared__ _Float16 tile[32][33];     // [c_local][l_local]
    const int tx = threadIdx.x & 31, ty = threadIdx.x >> 5;
    #pragma unroll
    for (int i = 0; i < 4; ++i)
        tile[ty + i*8][tx] = (_Float16)x[((size_t)b*Cc + c0 + ty + i*8)*Ll + l0 + tx];
    __syncthreads();
    #pragma unroll
    for (int i = 0; i < 4; ++i)
        xh[((size_t)b*Ll + l0 + ty + i*8)*Cc + c0 + tx] = tile[tx][ty + i*8];
}

// res [B*L,C] -> out [B,C,L], + x residual
__global__ __launch_bounds__(256)
void final_out(const float* __restrict__ res, const float* __restrict__ x,
               float* __restrict__ out)
{
    const int b = blockIdx.z, l0 = blockIdx.x*32, c0 = blockIdx.y*32;
    __shared__ float tile[32][33];        // [l_local][c_local]
    const int tx = threadIdx.x & 31, ty = threadIdx.x >> 5;
    #pragma unroll
    for (int i = 0; i < 4; ++i)
        tile[ty + i*8][tx] = res[((size_t)b*Ll + l0 + ty + i*8)*Cc + c0 + tx];
    __syncthreads();
    #pragma unroll
    for (int i = 0; i < 4; ++i) {
        const size_t o = ((size_t)b*Cc + c0 + ty + i*8)*Ll + l0 + tx;
        out[o] = tile[tx][ty + i*8] + x[o];
    }
}

__global__ void castw(const float* __restrict__ w, _Float16* __restrict__ h, int n)
{
    int i = blockIdx.x*256 + threadIdx.x;
    if (i < n) h[i] = (_Float16)w[i];
}

// q_probe: mean over L of Q[., h*64+d], chunked partials then reduce
__global__ __launch_bounds__(256)
void qprobe_part(const float* __restrict__ Q, float* __restrict__ qpp)
{
    const int bh = blockIdx.x, ch = blockIdx.y, b = bh >> 3, h = bh & 7;
    const int t = threadIdx.x, d = t & 63, p = t >> 6;
    const int l0 = ch * (Ll / 8);
    float s = 0.0f;
    for (int l = l0 + p; l < l0 + Ll/8; l += 4)
        s += Q[((size_t)b*Ll + l)*Cc + h*64 + d];
    __shared__ float sh[256];
    sh[t] = s; __syncthreads();
    if (p == 0)
        qpp[((size_t)bh*8 + ch)*64 + d] = sh[d] + sh[64+d] + sh[128+d] + sh[192+d];
}

__global__ void qprobe_red(const float* __restrict__ qpp, float* __restrict__ qp)
{
    const int bh = blockIdx.x, t = threadIdx.x;   // 64 threads
    float s = 0.0f;
    for (int ch = 0; ch < 8; ++ch) s += qpp[((size_t)bh*8 + ch)*64 + t];
    qp[bh*64 + t] = s / (float)Ll;
}

// score[bh*L + l] = dot(q_probe, K_row) / 8
__global__ __launch_bounds__(256)
void score_k(const float* __restrict__ qp, const float* __restrict__ Kf,
             float* __restrict__ score)
{
    const int idx = blockIdx.x*256 + threadIdx.x;     // B*H*L = 131072
    const int bh = idx >> 12, l = idx & (Ll - 1);
    const int b = bh >> 3, h = bh & 7;
    float s = 0.0f;
    #pragma unroll 8
    for (int d = 0; d < 64; ++d)
        s += qp[bh*64 + d] * Kf[((size_t)b*Ll + l)*Cc + h*64 + d];
    score[idx] = s * 0.125f;
}

// softmax stats over Lk = {zero-token} U L real scores
__global__ __launch_bounds__(256)
void stats_k(const float* __restrict__ score, float* __restrict__ smax,
             float* __restrict__ sinv)
{
    const int bh = blockIdx.x, t = threadIdx.x;
    const float* s = score + (size_t)bh * Ll;
    __shared__ float sh[256];
    float mx = 0.0f;                                 // zero-token score
    for (int l = t; l < Ll; l += 256) mx = fmaxf(mx, s[l]);
    sh[t] = mx; __syncthreads();
    for (int o = 128; o > 0; o >>= 1) { if (t < o) sh[t] = fmaxf(sh[t], sh[t+o]); __syncthreads(); }
    mx = sh[0]; __syncthreads();
    float sm = 0.0f;
    for (int l = t; l < Ll; l += 256) sm += expf(s[l] - mx);
    sh[t] = sm; __syncthreads();
    for (int o = 128; o > 0; o >>= 1) { if (t < o) sh[t] += sh[t+o]; __syncthreads(); }
    if (t == 0) { smax[bh] = mx; sinv[bh] = 1.0f / (sh[0] + expf(-mx)); }
}

// kv partials: per (bh, L-chunk) accumulate kv[64][64] and ksum[64]
__global__ __launch_bounds__(256)
void kv_part(const float* __restrict__ phiK, const float* __restrict__ Vf,
             const float* __restrict__ score, const float* __restrict__ smax,
             const float* __restrict__ sinv, float* __restrict__ kvp,
             float* __restrict__ kstp)
{
    const int bh = blockIdx.x, ch = blockIdx.y, b = bh >> 3, h = bh & 7;
    const int t = threadIdx.x, d = t & 63, eq = t >> 6;
    const float mx = smax[bh], inv = sinv[bh];
    float acc[16] = {};
    float ks = 0.0f;
    __shared__ float sp[64], sv[64];
    const int l0 = ch * (Ll / 32);
    for (int l = l0; l < l0 + Ll/32; ++l) {
        if (t < 64)        sp[t]    = phiK[((size_t)b*Ll + l)*Cc + h*64 + t];
        else if (t < 128)  sv[t-64] = Vf  [((size_t)b*Ll + l)*Cc + h*64 + (t - 64)];
        __syncthreads();
        const float wgt = expf(score[(size_t)bh*Ll + l] - mx) * inv;
        const float pw  = sp[d] * wgt;
        #pragma unroll
        for (int j = 0; j < 16; ++j) acc[j] += pw * sv[eq*16 + j];
        if (eq == 0) ks += pw;
        __syncthreads();
    }
    #pragma unroll
    for (int j = 0; j < 16; ++j)
        kvp[(((size_t)bh*32 + ch)*64 + d)*64 + eq*16 + j] = acc[j];
    if (eq == 0) kstp[((size_t)bh*32 + ch)*64 + d] = ks;
}

// reduce kv chunks -> kv_h (f16), ksum (+ analytic zero-token term)
__global__ __launch_bounds__(256)
void kv_red(const float* __restrict__ kvp, const float* __restrict__ kstp,
            const float* __restrict__ smax, const float* __restrict__ sinv,
            const float* __restrict__ bkk, _Float16* __restrict__ kv_h,
            float* __restrict__ ksum)
{
    const int bh = blockIdx.x, h = bh & 7;
    const int t = threadIdx.x, d = t & 63, eq = t >> 6;
    #pragma unroll
    for (int j = 0; j < 16; ++j) {
        float s = 0.0f;
        for (int ch = 0; ch < 32; ++ch)
            s += kvp[(((size_t)bh*32 + ch)*64 + d)*64 + eq*16 + j];
        kv_h[((size_t)bh*64 + d)*64 + eq*16 + j] = (_Float16)s;
    }
    if (eq == 0) {
        float s = 0.0f;
        for (int ch = 0; ch < 32; ++ch) s += kstp[((size_t)bh*32 + ch)*64 + d];
        const float w0 = expf(-smax[bh]) * sinv[bh];
        s += (tanhf(bkk[h*64 + d]) + 1.0f) * w0;    // zero-token contribution
        ksum[bh*64 + d] = s;
    }
}

// bottom[bh*L + l] = dot(phiQ_row, ksum) + 1e-6
__global__ __launch_bounds__(256)
void bottom_k(const float* __restrict__ phiQ, const float* __restrict__ ksum,
              float* __restrict__ bottom)
{
    const int idx = blockIdx.x*256 + threadIdx.x;
    const int bh = idx >> 12, l = idx & (Ll - 1);
    const int b = bh >> 3, h = bh & 7;
    float s = 0.0f;
    #pragma unroll 8
    for (int d = 0; d < 64; ++d)
        s += phiQ[((size_t)b*Ll + l)*Cc + h*64 + d] * ksum[bh*64 + d];
    bottom[idx] = s + 1e-6f;
}

// ---------------------------------------------------------------------------
extern "C" void kernel_launch(void* const* d_in, const int* in_sizes, int n_in,
                              void* d_out, int out_size, void* d_ws, size_t ws_size,
                              hipStream_t stream)
{
    (void)in_sizes; (void)n_in; (void)out_size; (void)ws_size;
    const float* x    = (const float*)d_in[0];
    const float* Wq   = (const float*)d_in[1];
    const float* bq   = (const float*)d_in[2];
    const float* gq   = (const float*)d_in[3];
    const float* btq  = (const float*)d_in[4];
    const float* Wk   = (const float*)d_in[5];
    const float* bk   = (const float*)d_in[6];
    const float* gk   = (const float*)d_in[7];
    const float* btk  = (const float*)d_in[8];
    const float* Wv   = (const float*)d_in[9];
    const float* bv   = (const float*)d_in[10];
    const float* gv   = (const float*)d_in[11];
    const float* btv  = (const float*)d_in[12];
    const float* Wkq  = (const float*)d_in[13];
    const float* bkq  = (const float*)d_in[14];
    const float* Wkk  = (const float*)d_in[15];
    const float* bkk  = (const float*)d_in[16];
    const float* gat  = (const float*)d_in[17];
    const float* bat  = (const float*)d_in[18];
    const float* W1   = (const float*)d_in[19];
    const float* b1   = (const float*)d_in[20];
    const float* W2   = (const float*)d_in[21];
    const float* b2   = (const float*)d_in[22];
    float* out = (float*)d_out;

    // ---- workspace carving -------------------------------------------------
    char* wsb = (char*)d_ws;
    size_t off = 0;
    auto carve = [&](size_t bytes) -> char* {
        char* p = wsb + off;
        off = (off + bytes + 255) & ~(size_t)255;
        return p;
    };
    const size_t NCf = (size_t)Nrow * Cc * sizeof(float);     // 33.5 MB
    const size_t NCh = (size_t)Nrow * Cc * sizeof(_Float16);  // 16.8 MB

    float* S0 = (float*)carve(NCf);            // Qf -> a
    float* S1 = (float*)carve(NCf);            // Kf -> attn_out -> res
    float* S3 = (float*)carve(NCf);            // phiQf
    float* S2 = (float*)carve(NCf);            // Vf   (S2,S4 contiguous -> h1h)
    float* S4 = (float*)carve(NCf);            // gemm tmp / phiKf
    _Float16* h1h   = (_Float16*)S2;           // [N, 2048] reuses S2+S4 span

    _Float16* xh    = (_Float16*)carve(NCh);
    _Float16* Qh    = (_Float16*)carve(NCh);   // -> a_h
    _Float16* Kh    = (_Float16*)carve(NCh);
    _Float16* phiQh = (_Float16*)carve(NCh);

    _Float16* Wqh  = (_Float16*)carve(Cc*Cc*2);
    _Float16* Wkh  = (_Float16*)carve(Cc*Cc*2);
    _Float16* Wvh  = (_Float16*)carve(Cc*Cc*2);
    _Float16* Wkqh = (_Float16*)carve(Cc*Cc*2);
    _Float16* Wkkh = (_Float16*)carve(Cc*Cc*2);
    _Float16* W1h  = (_Float16*)carve((size_t)Cc*MLPd*2);
    _Float16* W2h  = (_Float16*)carve((size_t)MLPd*Cc*2);

    float* score  = (float*)carve((size_t)Bb*Hh*Ll*4);
    float* bottom = (float*)carve((size_t)Bb*Hh*Ll*4);
    float* qpp    = (float*)carve((size_t)Bb*Hh*8*64*4);
    float* qp     = (float*)carve((size_t)Bb*Hh*64*4);
    float* smax   = (float*)carve(Bb*Hh*4);
    float* sinv   = (float*)carve(Bb*Hh*4);
    float* kvp    = (float*)carve((size_t)Bb*Hh*32*64*64*4);
    float* kstp   = (float*)carve((size_t)Bb*Hh*32*64*4);
    _Float16* kvh = (_Float16*)carve((size_t)Bb*Hh*64*64*2);
    float* ksum   = (float*)carve((size_t)Bb*Hh*64*4);

    // ---- weight casts + input transpose ------------------------------------
    castw<<<(Cc*Cc+255)/256,     256, 0, stream>>>(Wq,  Wqh,  Cc*Cc);
    castw<<<(Cc*Cc+255)/256,     256, 0, stream>>>(Wk,  Wkh,  Cc*Cc);
    castw<<<(Cc*Cc+255)/256,     256, 0, stream>>>(Wv,  Wvh,  Cc*Cc);
    castw<<<(Cc*Cc+255)/256,     256, 0, stream>>>(Wkq, Wkqh, Cc*Cc);
    castw<<<(Cc*Cc+255)/256,     256, 0, stream>>>(Wkk, Wkkh, Cc*Cc);
    castw<<<(Cc*MLPd+255)/256,   256, 0, stream>>>(W1,  W1h,  Cc*MLPd);
    castw<<<(MLPd*Cc+255)/256,   256, 0, stream>>>(W2,  W2h,  MLPd*Cc);
    xpose_in<<<dim3(Ll/32, Cc/32, Bb), 256, 0, stream>>>(x, xh);

    const dim3 gFull(Nrow/128, Cc/128, 1);     // 128 x 4, tile 128x128

    // ---- Q, K, V: GEMM + LN -------------------------------------------------
    gemm_wmma<0,128,4><<<gFull, 256, 0, stream>>>(xh, Wqh, S4, nullptr, bq, nullptr, nullptr,
        Cc, Cc, Cc, Nrow, Cc, 1, 0,0,0,0,0,0,0);
    ln_rows<<<Nrow, 256, 0, stream>>>(S4, gq, btq, S0, Qh);

    gemm_wmma<0,128,4><<<gFull, 256, 0, stream>>>(xh, Wkh, S4, nullptr, bk, nullptr, nullptr,
        Cc, Cc, Cc, Nrow, Cc, 1, 0,0,0,0,0,0,0);
    ln_rows<<<Nrow, 256, 0, stream>>>(S4, gk, btk, S1, Kh);

    gemm_wmma<0,128,4><<<gFull, 256, 0, stream>>>(xh, Wvh, S4, nullptr, bv, nullptr, nullptr,
        Cc, Cc, Cc, Nrow, Cc, 1, 0,0,0,0,0,0,0);
    ln_rows<<<Nrow, 256, 0, stream>>>(S4, gv, btv, S2, nullptr);

    // ---- phiQ, phiK ---------------------------------------------------------
    gemm_wmma<1,128,4><<<gFull, 256, 0, stream>>>(Qh, Wkqh, S3, phiQh, bkq, nullptr, nullptr,
        Cc, Cc, Cc, Nrow, Cc, 1, 0,0,0,0,0,0,0);
    gemm_wmma<1,128,4><<<gFull, 256, 0, stream>>>(Kh, Wkkh, S4, nullptr, bkk, nullptr, nullptr,
        Cc, Cc, Cc, Nrow, Cc, 1, 0,0,0,0,0,0,0);

    // ---- attention reductions ----------------------------------------------
    qprobe_part<<<dim3(Bb*Hh, 8), 256, 0, stream>>>(S0, qpp);
    qprobe_red<<<Bb*Hh, 64, 0, stream>>>(qpp, qp);
    score_k<<<(Bb*Hh*Ll)/256, 256, 0, stream>>>(qp, S1, score);
    stats_k<<<Bb*Hh, 256, 0, stream>>>(score, smax, sinv);
    kv_part<<<dim3(Bb*Hh, 32), 256, 0, stream>>>(S4, S2, score, smax, sinv, kvp, kstp);
    kv_red<<<Bb*Hh, 256, 0, stream>>>(kvp, kstp, smax, sinv, bkk, kvh, ksum);
    bottom_k<<<(Bb*Hh*Ll)/256, 256, 0, stream>>>(S3, ksum, bottom);

    // ---- num = phiQ @ kv, / bottom  (batched over b,h) ----------------------
    gemm_wmma<3,64,8><<<dim3(Ll/128, 1, Bb*Hh), 256, 0, stream>>>(
        phiQh, kvh, S1, nullptr, nullptr, nullptr, bottom,
        Cc, 64, Cc, Ll, 64, Hh,
        (long long)Ll*Cc, 64,                 // A strides (b, h)
        (long long)Hh*64*64, 64*64,           // B strides (b, h)
        (long long)Ll*Cc, 64,                 // out strides (b, h)
        Ll);                                  // rowdiv stride per batch

    // ---- attn LN, MLP, residual --------------------------------------------
    ln_rows<<<Nrow, 256, 0, stream>>>(S1, gat, bat, S0, Qh);     // a = S0, a_h = Qh

    gemm_wmma<2,128,4><<<dim3(Nrow/128, MLPd/128, 1), 256, 0, stream>>>(
        Qh, W1h, nullptr, h1h, b1, nullptr, nullptr,
        Cc, MLPd, MLPd, Nrow, Cc, 1, 0,0,0,0,0,0,0);

    gemm_wmma<4,128,4><<<gFull, 256, 0, stream>>>(
        h1h, W2h, S1, nullptr, b2, S0, nullptr,
        MLPd, Cc, Cc, Nrow, MLPd, 1, 0,0,0,0,0,0,0);

    final_out<<<dim3(Ll/32, Cc/32, Bb), 256, 0, stream>>>(S1, x, out);
}
